// SpikingCNN_85315230367870
// MI455X (gfx1250) — matrix operand order repaired and checked by reading, not compiled, not run on hardware
//
#include <hip/hip_runtime.h>

typedef __attribute__((ext_vector_type(16))) _Float16 v16h;
typedef __attribute__((ext_vector_type(8)))  float    v8f;

namespace {
constexpr int   T_     = 15, CIN = 6, HIN = 512, WIN_ = 512;
constexpr int   F_     = 64, KSZ = 5, PADC = 2;
constexpr int   HO     = 256, HP = HO + 2 * PADC;      // 260 rows
constexpr int   HPW    = 264;                          // padded row width (16B-aligned stride)
constexpr int   KDIM   = CIN * KSZ * KSZ;              // 150
constexpr float THRESH = 15.0f;
constexpr int   KWTA   = 5, RADIUS = 2;
constexpr int   NPIX   = HO * HO;                      // 65536
constexpr size_t SPK_ELEMS = (size_t)T_ * F_ * HO * HO;

constexpr size_t alignUp(size_t v, size_t a) { return (v + a - 1) & ~(a - 1); }
constexpr size_t IN_ELEMS = (size_t)T_ * CIN * HP * HPW;   // pooled+padded f16
constexpr size_t W_ELEMS  = 5 * 4 * 32 * 16;               // [ks][mt][lane][16] f16
constexpr size_t OFF_IN   = 0;
constexpr size_t OFF_W    = alignUp(IN_ELEMS * 2, 256);
constexpr size_t OFF_RECW = alignUp(OFF_W + W_ELEMS * 2, 256);
constexpr size_t OFF_RECC = OFF_RECW + (size_t)NPIX * 4;
constexpr size_t OFF_RECV = OFF_RECC + (size_t)NPIX * 4;
constexpr size_t OFF_VMAX = alignUp(OFF_RECV + (size_t)NPIX * 4, 256);
constexpr size_t OFF_WREC = OFF_VMAX + 256;                // 5 * 4 ints
constexpr size_t OFF_PVAL = OFF_WREC + 256;                // 256 floats
constexpr size_t OFF_PIDX = OFF_PVAL + 1024;               // 256 ints

constexpr int NROWS   = CIN * KSZ;      // 30 halo rows per block
constexpr int ROWH    = 144;            // halo row width in halves (288 B, 16B mult)
constexpr int WCHUNKS = (int)(W_ELEMS * 2 / 16);     // 1280
constexpr int ICHUNKS = NROWS * (ROWH * 2 / 16);     // 30*18 = 540
}

__device__ __forceinline__ void async_ld_b128(unsigned ldsAddr, unsigned long long gAddr) {
  asm volatile("global_load_async_to_lds_b128 %0, %1, off"
               :: "v"(ldsAddr), "v"(gAddr)
               : "memory");
}
__device__ __forceinline__ void wait_async0() {
  asm volatile("s_wait_asynccnt 0" ::: "memory");
}

// ---- 1) 2x2 max-pool (stride 2) + zero-pad(2), f16 [T][CIN][260][264]
__global__ __launch_bounds__(256) void pool_pad_kernel(
    const float* __restrict__ x, _Float16* __restrict__ pin)
{
  const size_t idx = (size_t)blockIdx.x * blockDim.x + threadIdx.x;
  if (idx >= IN_ELEMS) return;
  const int xx = (int)(idx % HPW);
  size_t r = idx / HPW;
  const int yy = (int)(r % HP); r /= HP;
  const int c  = (int)(r % CIN);
  const int t  = (int)(r / CIN);
  float m = 0.0f;
  if (yy >= PADC && yy < HP - PADC && xx >= PADC && xx < HP - PADC - 2) { // data cols 2..255
    const int iy = (yy - PADC) * 2, ix = (xx - PADC) * 2;
    const float* p = x + (((size_t)t * CIN + c) * HIN + iy) * WIN_ + ix;
    m = fmaxf(fmaxf(p[0], p[1]), fmaxf(p[WIN_], p[WIN_ + 1]));
  }
  pin[idx] = (_Float16)m;
}

// ---- 2) repack weights into WMMA 16-bit A-fragment lane layout (K padded to 160)
__global__ __launch_bounds__(256) void repack_w_kernel(
    const float* __restrict__ w, _Float16* __restrict__ wrep)
{
  const int idx = blockIdx.x * blockDim.x + threadIdx.x;
  if (idx >= (int)W_ELEMS) return;
  const int j    = idx & 15;
  const int lane = (idx >> 4) & 31;
  const int mt   = (idx >> 9) & 3;
  const int ks   = idx >> 11;
  // A in this code's contiguous k order (B matches): lanes0-15 K = j; lanes16-31 K = j+16
  const int kLocal = j + ((lane >= 16) ? 16 : 0);
  const int Kg = ks * 32 + kLocal;
  const int f  = mt * 16 + (lane & 15);
  float v = 0.0f;
  if (Kg < KDIM) {
    const int ci = Kg / 25, rem = Kg - ci * 25;
    const int kr = rem / 5, kc = rem - kr * 5;
    v = w[(((size_t)f * CIN + ci) * KSZ + kr) * KSZ + kc];
  }
  wrep[idx] = (_Float16)v;
}

// ---- 3) implicit-GEMM conv: async global->LDS staging, LDS-fed WMMA, fire
// block = 8 waves = 128 pixels of one output row; wave = 16 pixels x 64 features
__global__ __launch_bounds__(256) void conv_wmma_kernel(
    const _Float16* __restrict__ in, const _Float16* __restrict__ wrep,
    float* __restrict__ out)
{
  __shared__ _Float16 sW[W_ELEMS];          // 20480 B
  __shared__ _Float16 sIn[NROWS * ROWH];    // 8640 B

  const int tid  = threadIdx.x;
  const int lane = tid & 31;
  const int wib  = tid >> 5;                 // wave in block
  const int xblk = blockIdx.x & 1;
  const int oy   = (blockIdx.x >> 1) & 255;
  const int t    = blockIdx.x >> 9;
  const int oxb  = xblk * 128;
  const int n    = lane & 15;
  const int hi   = (lane >= 16) ? 1 : 0;
  const int xl   = wib * 16 + n;             // local col 0..127

  // --- async stage: weights (1280 x 16B) + input halo (540 x 16B)
  {
    const unsigned swBase = (unsigned)(size_t)&sW[0];
    const unsigned long long gwBase = (unsigned long long)(size_t)wrep;
    for (int ch = tid; ch < WCHUNKS; ch += 256)
      async_ld_b128(swBase + ch * 16, gwBase + (unsigned long long)ch * 16);

    const unsigned siBase = (unsigned)(size_t)&sIn[0];
    const unsigned long long giBase =
        (unsigned long long)(size_t)(in + (size_t)t * CIN * HP * HPW);
    for (int ch = tid; ch < ICHUNKS; ch += 256) {
      const int row = ch / 18, c16 = ch % 18;       // row = ci*5 + kr
      const int ci = row / 5, kr = row - ci * 5;
      const unsigned long long g =
          giBase + ((unsigned long long)((ci * HP + (oy + kr)) * HPW + oxb)) * 2
                 + (unsigned long long)c16 * 16;
      async_ld_b128(siBase + (unsigned)(row * ROWH * 2 + c16 * 16), g);
    }
  }
  wait_async0();
  __syncthreads();

  const v16h* __restrict__ sWv = (const v16h*)sW;
  v8f acc[4] = {};
#pragma unroll
  for (int ks = 0; ks < 5; ++ks) {
    v16h a[4];
#pragma unroll
    for (int mt = 0; mt < 4; ++mt)
      a[mt] = sWv[(ks * 4 + mt) * 32 + lane];       // 2x ds_load_b128 each

    // B 32x16: lanes0-15 hold K 0..15, lanes16-31 hold K 16..31
    v16h b;
    if (!hi) {
#pragma unroll
      for (int j = 0; j < 16; ++j) {
        const int Kg = ks * 32 + j;
        _Float16 v = (_Float16)0.0f;
        if (Kg < KDIM) {
          const int ci = Kg / 25, rem = Kg - ci * 25;
          const int kr = rem / 5,  kc = rem - kr * 5;
          v = sIn[(ci * 5 + kr) * ROWH + xl + kc];  // ds_load_u16, imm offset
        }
        b[j] = v;
      }
    } else {
#pragma unroll
      for (int j = 0; j < 16; ++j) {
        const int Kg = ks * 32 + 16 + j;
        _Float16 v = (_Float16)0.0f;
        if (Kg < KDIM) {
          const int ci = Kg / 25, rem = Kg - ci * 25;
          const int kr = rem / 5,  kc = rem - kr * 5;
          v = sIn[(ci * 5 + kr) * ROWH + xl + kc];
        }
        b[j] = v;
      }
    }

#pragma unroll
    for (int mt = 0; mt < 4; ++mt)
      acc[mt] = __builtin_amdgcn_wmma_f32_16x16x32_f16(
          false, a[mt], false, b, (short)0, acc[mt], false, false);
  }

  // C layout: VGPR e -> M = e (lanes0-15) / e+8 (lanes16-31); N = lane&15
  const size_t spat = (size_t)oy * HO + (oxb + xl - xl % 16) + n; // = oy*HO + oxb + wib*16 + n
#pragma unroll
  for (int mt = 0; mt < 4; ++mt) {
#pragma unroll
    for (int e = 0; e < 8; ++e) {
      const int f = mt * 16 + e + hi * 8;
      float p = acc[mt][e];
      p = (p < THRESH) ? 0.0f : p;                 // fire
      out[((size_t)t * F_ + f) * NPIX + spat] = p; // pot (binarized later)
    }
  }
}

// ---- 4) pointwise inhibition: rewrite spk in place, emit per-pixel k-WTA record
__global__ __launch_bounds__(256) void inhibit_kernel(
    float* __restrict__ out, int* __restrict__ recW,
    float* __restrict__ recC, float* __restrict__ recV,
    unsigned int* __restrict__ vmax)
{
  const int pix = blockIdx.x * blockDim.x + threadIdx.x;
  if (pix >= NPIX) return;
  float* __restrict__ base = out + pix;     // stride NPIX per (t,f)

  int cnt = 0, lastC = 0;
  for (int t = 0; t < T_; ++t) {
    float m = 0.0f;
    for (int f = 0; f < F_; ++f)
      m = fmaxf(m, base[(size_t)(t * F_ + f) * NPIX]);
    const int c = (m > 0.0f) ? 1 : 0;
    cnt += c;
    if (t == T_ - 1) lastC = c;
  }
  int earliest = T_ - cnt;
  earliest = earliest < 0 ? 0 : (earliest > T_ - 1 ? T_ - 1 : earliest);

  int winner = 0;
  float wv = base[(size_t)(earliest * F_) * NPIX];
  for (int f = 1; f < F_; ++f) {
    const float p = base[(size_t)(earliest * F_ + f) * NPIX];
    if (p > wv) { wv = p; winner = f; }          // first-max tie-break
  }

  float pw[T_];
  int cnt2 = 0;
  for (int t = 0; t < T_; ++t) {
    pw[t] = base[(size_t)(t * F_ + winner) * NPIX];
    cnt2 += (pw[t] > 0.0f) ? 1 : 0;
  }
  int e2 = T_ - cnt2;
  e2 = e2 < 0 ? 0 : (e2 > T_ - 1 ? T_ - 1 : e2);
  const float valw = pw[e2];
  const int act = (lastC && cnt2 > 0) ? 1 : 0;

  recW[pix] = act ? winner : -1;
  recC[pix] = act ? (float)cnt2 : 0.0f;
  recV[pix] = act ? valw : 0.0f;
  if (act) atomicMax(vmax, __float_as_uint(valw));  // valw >= 0

  for (int t = 0; t < T_; ++t) {
    const float sw = (act && pw[t] > 0.0f) ? 1.0f : 0.0f;
    for (int f = 0; f < F_; ++f)
      base[(size_t)(t * F_ + f) * NPIX] = (f == winner) ? sw : 0.0f;
  }
}

// ---- 5) k-WTA: grid reduce over per-pixel records with suppression masking
__global__ __launch_bounds__(256) void kwta_scan_kernel(
    const int* __restrict__ recW, const float* __restrict__ recC,
    const float* __restrict__ recV, const unsigned int* __restrict__ vmax,
    const int* __restrict__ wrec, int round,
    float* __restrict__ pval, int* __restrict__ pidx)
{
  const int pix = blockIdx.x * 256 + threadIdx.x;
  const int y = pix >> 8, xc = pix & 255;
  const float v = __uint_as_float(*vmax) * (float)T_;

  float score = 0.0f;
  int flat = 0x7fffffff;
  const int wf = recW[pix];
  const float c = recC[pix];
  if (wf >= 0 && c > 0.0f) {
    score = c * (recV[pix] + v);
    flat = wf * NPIX + pix;
    for (int r2 = 0; r2 < round; ++r2) {
      const int pf = wrec[r2 * 4 + 0], py = wrec[r2 * 4 + 1];
      const int px = wrec[r2 * 4 + 2], pa = wrec[r2 * 4 + 3];
      if (pa && (wf == pf || (abs(y - py) <= RADIUS && abs(xc - px) <= RADIUS))) {
        score = 0.0f;
        break;
      }
    }
  }

  __shared__ float sv[256];
  __shared__ int   si[256];
  sv[threadIdx.x] = score; si[threadIdx.x] = flat;
  for (int s = 128; s > 0; s >>= 1) {
    __syncthreads();
    if ((int)threadIdx.x < s) {
      const float ov = sv[threadIdx.x + s];
      const int   oi = si[threadIdx.x + s];
      if (ov > sv[threadIdx.x] ||
          (ov == sv[threadIdx.x] && oi < si[threadIdx.x])) {
        sv[threadIdx.x] = ov; si[threadIdx.x] = oi;
      }
    }
  }
  if (threadIdx.x == 0) { pval[blockIdx.x] = sv[0]; pidx[blockIdx.x] = si[0]; }
}

__global__ __launch_bounds__(256) void kwta_final_kernel(
    const float* __restrict__ pval, const int* __restrict__ pidx,
    int* __restrict__ wrec, int round, float* __restrict__ outWin)
{
  __shared__ float sv[256];
  __shared__ int   si[256];
  sv[threadIdx.x] = pval[threadIdx.x];
  si[threadIdx.x] = pidx[threadIdx.x];
  for (int s = 128; s > 0; s >>= 1) {
    __syncthreads();
    if ((int)threadIdx.x < s) {
      const float ov = sv[threadIdx.x + s];
      const int   oi = si[threadIdx.x + s];
      if (ov > sv[threadIdx.x] ||
          (ov == sv[threadIdx.x] && oi < si[threadIdx.x])) {
        sv[threadIdx.x] = ov; si[threadIdx.x] = oi;
      }
    }
  }
  if (threadIdx.x == 0) {
    const float val = sv[0];
    const int idx = si[0];
    const int active = (val != 0.0f && idx != 0x7fffffff) ? 1 : 0;
    int f = -1, yy = -1, xx = -1;
    if (active) { f = idx >> 16; yy = (idx >> 8) & 255; xx = idx & 255; }
    wrec[round * 4 + 0] = f;  wrec[round * 4 + 1] = yy;
    wrec[round * 4 + 2] = xx; wrec[round * 4 + 3] = active;
    outWin[round * 3 + 0] = (float)f;
    outWin[round * 3 + 1] = (float)yy;
    outWin[round * 3 + 2] = (float)xx;
  }
}

extern "C" void kernel_launch(void* const* d_in, const int* in_sizes, int n_in,
                              void* d_out, int out_size, void* d_ws, size_t ws_size,
                              hipStream_t stream)
{
  const float* x = (const float*)d_in[0];
  const float* w = (const float*)d_in[1];
  float* out = (float*)d_out;
  char* ws = (char*)d_ws;

  _Float16* pin  = (_Float16*)(ws + OFF_IN);
  _Float16* wrep = (_Float16*)(ws + OFF_W);
  int*      recW = (int*)(ws + OFF_RECW);
  float*    recC = (float*)(ws + OFF_RECC);
  float*    recV = (float*)(ws + OFF_RECV);
  unsigned* vmax = (unsigned*)(ws + OFF_VMAX);
  int*      wrec = (int*)(ws + OFF_WREC);
  float*    pval = (float*)(ws + OFF_PVAL);
  int*      pidx = (int*)(ws + OFF_PIDX);

  hipMemsetAsync(vmax, 0, sizeof(unsigned), stream);
  pool_pad_kernel<<<(int)((IN_ELEMS + 255) / 256), 256, 0, stream>>>(x, pin);
  repack_w_kernel<<<(int)((W_ELEMS + 255) / 256), 256, 0, stream>>>(w, wrep);
  // 15*256*2 = 7680 blocks, 8 waves/block, 128 pixels/block
  conv_wmma_kernel<<<7680, 256, 0, stream>>>(pin, wrep, out);
  inhibit_kernel<<<NPIX / 256, 256, 0, stream>>>(out, recW, recC, recV, vmax);
  for (int r = 0; r < KWTA; ++r) {
    kwta_scan_kernel<<<256, 256, 0, stream>>>(recW, recC, recV, vmax, wrec, r, pval, pidx);
    kwta_final_kernel<<<1, 256, 0, stream>>>(pval, pidx, wrec, r, out + SPK_ELEMS);
  }
}
